// MPNN_80187039416386
// MI455X (gfx1250) — matrix operand order repaired and checked by reading, not compiled, not run on hardware
//
#include <hip/hip_runtime.h>
#include <hip/hip_bf16.h>

// ---------------- problem constants (match reference) ----------------
#define N_NODES    50000
#define N_EDGES    800000
#define NODE_DIM   32
#define EDGE_DIM   16
#define HID        64
#define MSG_HID    128
#define NUM_GRAPHS 512
#define NUM_MP     3
#define S2S_STEPS  6

typedef __attribute__((ext_vector_type(16))) _Float16 v16h;
typedef __attribute__((ext_vector_type(8)))  _Float16 v8h;
typedef __attribute__((ext_vector_type(8)))  float    v8f;

static __device__ __forceinline__ v16h mk16(v8h lo, v8h hi) {
  v16h r;
#pragma unroll
  for (int i = 0; i < 8; ++i) { r[i] = lo[i]; r[i + 8] = hi[i]; }
  return r;
}

static __device__ __forceinline__ v8f wmmaf16(v16h a, v16h b, v8f c) {
  return __builtin_amdgcn_wmma_f32_16x16x32_f16(false, a, false, b, (short)0, c, false, false);
}

// load a pre-packed B fragment: 32 contiguous bytes per lane (L2 broadcast)
static __device__ __forceinline__ v16h ldbfrag(const _Float16* __restrict__ P, int fid, int lane) {
  const v8h* bp = (const v8h*)(P + ((size_t)(fid * 32 + lane) << 4));
  return mk16(bp[0], bp[1]);
}

// load an A fragment half-pair directly from a row-major f16 row:
// lane m = lane&15 owns row; K-halves are contiguous 16-B slices.
static __device__ __forceinline__ v16h ldafrag(const _Float16* __restrict__ row, int kc, int half) {
  v8h lo = *(const v8h*)(row + kc * 32 + half * 8);
  v8h hi = *(const v8h*)(row + kc * 32 + 16 + half * 8);
  return mk16(lo, hi);
}

static __device__ __forceinline__ float sigmf(float x) { return 1.f / (1.f + expf(-x)); }

// order-preserving float <-> uint mapping for atomicMax over signed floats
static __device__ __forceinline__ unsigned fmapu(float f) {
  unsigned u = __float_as_uint(f);
  return (u & 0x80000000u) ? ~u : (u | 0x80000000u);
}
static __device__ __forceinline__ float fumap(unsigned u) {
  return (u & 0x80000000u) ? __uint_as_float(u & 0x7fffffffu) : __uint_as_float(~u);
}

// ---------------- generic helpers ----------------
__global__ void zero_kernel(float* p, int n) {
  int i = blockIdx.x * 256 + threadIdx.x;
  if (i < n) p[i] = 0.f;
}

__global__ void f32_to_f16_kernel(const float* __restrict__ in, _Float16* __restrict__ out, int n) {
  int i = blockIdx.x * 256 + threadIdx.x;
  if (i < n) out[i] = (_Float16)in[i];
}

// ---------------- weight packing into WMMA B-fragment layout ----------------
// Fragment element (fid=kc*NT+nt, lane, j) holds B[kc*32 + (j>>3)*16 + (lane>>4)*8 + (j&7)][nt*16 + (lane&15)]
__global__ void pack_w1_kernel(const float* __restrict__ W, _Float16* __restrict__ out) {
  int idx = blockIdx.x * 256 + threadIdx.x;           // 5*8*32*16 = 20480
  if (idx >= 5 * 8 * 32 * 16) return;
  int j    = idx & 15;
  int lane = (idx >> 4) & 31;
  int nt   = (idx >> 9) & 7;
  int kc   = idx >> 12;
  int K = kc * 32 + (j >> 3) * 16 + (lane >> 4) * 8 + (j & 7); // padded K dim: 160
  int N = nt * 16 + (lane & 15);
  out[idx] = (K < (2 * HID + EDGE_DIM)) ? (_Float16)W[K * MSG_HID + N] : (_Float16)0.f;
}

__global__ void pack_w2_kernel(const float* __restrict__ W, _Float16* __restrict__ out) {
  int idx = blockIdx.x * 256 + threadIdx.x;           // 4*4*32*16 = 8192
  if (idx >= 4 * 4 * 32 * 16) return;
  int j    = idx & 15;
  int lane = (idx >> 4) & 31;
  int nt   = (idx >> 9) & 3;
  int kc   = idx >> 11;
  int K = kc * 32 + (j >> 3) * 16 + (lane >> 4) * 8 + (j & 7); // K = 128
  int N = nt * 16 + (lane & 15);
  out[idx] = (_Float16)W[K * HID + N];
}

// GRU weights used as m @ W^T: pack B[k][o] = W[o*64 + k], K=64 (2 chunks), N=192 (12 tiles)
__global__ void pack_gru_kernel(const float* __restrict__ W, _Float16* __restrict__ out) {
  int idx = blockIdx.x * 256 + threadIdx.x;           // 2*12*32*16 = 12288
  if (idx >= 2 * 12 * 32 * 16) return;
  int j    = idx & 15;
  int lane = (idx >> 4) & 31;
  int fid  = idx >> 9;                                // 0..23
  int nt   = fid % 12;
  int kc   = fid / 12;
  int K = kc * 32 + (j >> 3) * 16 + (lane >> 4) * 8 + (j & 7); // 0..63
  int N = nt * 16 + (lane & 15);                      // 0..191
  out[idx] = (_Float16)W[N * HID + K];
}

// ---------------- node embedding: x = x_feat @ W_emb + b ----------------
__global__ __launch_bounds__(256) void embed_kernel(
    const float* __restrict__ xf, const float* __restrict__ W, const float* __restrict__ b,
    float* __restrict__ h, _Float16* __restrict__ x16, int nNodes) {
  __shared__ float sW[NODE_DIM * HID];
  __shared__ float sX[4][NODE_DIM];
  int t = threadIdx.x;
  for (int i = t; i < NODE_DIM * HID; i += 256) sW[i] = W[i];
  int node0 = blockIdx.x * 4;
  if (t < 4 * NODE_DIM) {
    int nn = node0 + (t >> 5);
    sX[t >> 5][t & 31] = (nn < nNodes) ? xf[(size_t)nn * NODE_DIM + (t & 31)] : 0.f;
  }
  __syncthreads();
  int nl = t >> 6, c = t & 63;
  int node = node0 + nl;
  if (node < nNodes) {
    float s = b[c];
#pragma unroll
    for (int k = 0; k < NODE_DIM; ++k) s += sX[nl][k] * sW[k * HID + c];
    h[(size_t)node * HID + c] = s;
    x16[(size_t)node * HID + c] = (_Float16)s;
  }
}

// ---------------- edge MLP (WMMA) + scatter-add ----------------
// Wave handles 16 edges. A = [16,160] f16 (x[dst]|x[src]|edge_attr|pad) loaded as
// 16-B row slices straight from global (L2-resident node table) -- no staging LDS.
// K-outer loops with batched B-fragment preloads so loads clause together and
// pipeline against 8/4 back-to-back WMMAs (held f32 accumulators).
__global__ __launch_bounds__(128) void edge_mlp_kernel(
    const _Float16* __restrict__ x16, const _Float16* __restrict__ ea16,
    const int* __restrict__ srcArr, const int* __restrict__ dstArr,
    const _Float16* __restrict__ W1p, const _Float16* __restrict__ W2p,
    const float* __restrict__ b1, const float* __restrict__ b2,
    float* __restrict__ mbuf) {
  __shared__ _Float16 sH[4][16 * 128];   // 16 KB

  const int w    = threadIdx.x >> 5;
  const int lane = threadIdx.x & 31;
  const int tile = blockIdx.x * 4 + w;
  const int eb   = tile * 16;
  const int n    = lane & 15;            // A-row m / D-column n
  const int half = lane >> 4;

  // A fragments: K 0..63 = x[dst], 64..127 = x[src], 128..143 = edge_attr, 144..159 = 0
  const int e = eb + n;
  const _Float16* drow = x16 + (size_t)dstArr[e] * HID;
  const _Float16* srow = x16 + (size_t)srcArr[e] * HID;
  v16h afrag[5];
#pragma unroll
  for (int kc = 0; kc < 2; ++kc) afrag[kc]     = ldafrag(drow, kc, half);
#pragma unroll
  for (int kc = 0; kc < 2; ++kc) afrag[2 + kc] = ldafrag(srow, kc, half);
  {
    v8h lo = *(const v8h*)(ea16 + (size_t)e * EDGE_DIM + half * 8);
    v8h zz;
#pragma unroll
    for (int i = 0; i < 8; ++i) zz[i] = (_Float16)0.f;
    afrag[4] = mk16(lo, zz);
  }

  const v8f vzero = {};

  // GEMM1: [16,160] x [160,128] -> relu -> sH (f16)
  v8f acc[8];
#pragma unroll
  for (int nt = 0; nt < 8; ++nt) acc[nt] = vzero;
#pragma unroll
  for (int kc = 0; kc < 5; ++kc) {
    v16h bf[8];
#pragma unroll
    for (int nt = 0; nt < 8; ++nt) bf[nt] = ldbfrag(W1p, kc * 8 + nt, lane);
#pragma unroll
    for (int nt = 0; nt < 8; ++nt) acc[nt] = wmmaf16(afrag[kc], bf[nt], acc[nt]);
  }
#pragma unroll
  for (int nt = 0; nt < 8; ++nt) {
    float bias = b1[nt * 16 + n];
#pragma unroll
    for (int r = 0; r < 8; ++r) {
      float v = acc[nt][r] + bias;
      v = v > 0.f ? v : 0.f;
      sH[w][(r + half * 8) * 128 + nt * 16 + n] = (_Float16)v;
    }
  }
  __syncthreads();   // grid is exact (800000 % 64 == 0): uniform barrier

  // A2 fragments from sH (per-wave private tile)
  const _Float16* hrow = &sH[w][n * 128];
  v16h a2[4];
#pragma unroll
  for (int kc = 0; kc < 4; ++kc) a2[kc] = ldafrag(hrow, kc, half);

  int drowIdx[8];
#pragma unroll
  for (int r = 0; r < 8; ++r) drowIdx[r] = dstArr[eb + r + 8 * half];

  // GEMM2: [16,128] x [128,64] -> scatter-add into mbuf
  v8f acc2[4];
#pragma unroll
  for (int nt = 0; nt < 4; ++nt) acc2[nt] = vzero;
#pragma unroll
  for (int kc = 0; kc < 4; ++kc) {
    v16h bf[4];
#pragma unroll
    for (int nt = 0; nt < 4; ++nt) bf[nt] = ldbfrag(W2p, kc * 4 + nt, lane);
#pragma unroll
    for (int nt = 0; nt < 4; ++nt) acc2[nt] = wmmaf16(a2[kc], bf[nt], acc2[nt]);
  }
#pragma unroll
  for (int nt = 0; nt < 4; ++nt) {
    float bias = b2[nt * 16 + n];
#pragma unroll
    for (int r = 0; r < 8; ++r)
      atomicAdd(mbuf + (size_t)drowIdx[r] * HID + nt * 16 + n, acc2[nt][r] + bias);
  }
}

// ---------------- GRU cell via WMMA (torch gate order r,z,n) ----------------
// gi = m @ Wih^T, gh = h @ Whh^T as [16,64]x[64,192] WMMA GEMMs; gates on f32 accums.
__global__ __launch_bounds__(128) void gru_wmma_kernel(
    const _Float16* __restrict__ m16, float* __restrict__ h, _Float16* __restrict__ x16,
    const _Float16* __restrict__ Wihp, const _Float16* __restrict__ Whhp,
    const float* __restrict__ bih, const float* __restrict__ bhh, int nTiles) {
  const int w    = threadIdx.x >> 5;
  const int lane = threadIdx.x & 31;
  const int tile = blockIdx.x * 4 + w;
  if (tile >= nTiles) return;          // wave-uniform: EXEC stays all-ones for WMMA
  const int nb   = tile * 16;
  const int n    = lane & 15;
  const int half = lane >> 4;

  const _Float16* mrow = m16 + (size_t)(nb + n) * HID;
  const _Float16* hrow = x16 + (size_t)(nb + n) * HID;
  v16h am[2], ah[2];
#pragma unroll
  for (int kc = 0; kc < 2; ++kc) {
    am[kc] = ldafrag(mrow, kc, half);
    ah[kc] = ldafrag(hrow, kc, half);
  }

  const v8f vzero = {};

#pragma unroll
  for (int t = 0; t < 4; ++t) {        // 4 column tiles of HID
    // g6 = {ir, iz, in, hr, hz, hn} accumulators
    v8f g6[6];
#pragma unroll
    for (int q = 0; q < 6; ++q) g6[q] = vzero;
#pragma unroll
    for (int kc = 0; kc < 2; ++kc) {
      v16h bf[6];
      bf[0] = ldbfrag(Wihp, kc * 12 + t,     lane);
      bf[1] = ldbfrag(Wihp, kc * 12 + t + 4, lane);
      bf[2] = ldbfrag(Wihp, kc * 12 + t + 8, lane);
      bf[3] = ldbfrag(Whhp, kc * 12 + t,     lane);
      bf[4] = ldbfrag(Whhp, kc * 12 + t + 4, lane);
      bf[5] = ldbfrag(Whhp, kc * 12 + t + 8, lane);
      g6[0] = wmmaf16(am[kc], bf[0], g6[0]);
      g6[1] = wmmaf16(am[kc], bf[1], g6[1]);
      g6[2] = wmmaf16(am[kc], bf[2], g6[2]);
      g6[3] = wmmaf16(ah[kc], bf[3], g6[3]);
      g6[4] = wmmaf16(ah[kc], bf[4], g6[4]);
      g6[5] = wmmaf16(ah[kc], bf[5], g6[5]);
    }
    int c = t * 16 + n;
    float bir = bih[c], biz = bih[64 + c], bin = bih[128 + c];
    float bhr = bhh[c], bhz = bhh[64 + c], bhn = bhh[128 + c];
#pragma unroll
    for (int r = 0; r < 8; ++r) {
      int row = nb + r + 8 * half;
      float rg = sigmf((g6[0][r] + bir) + (g6[3][r] + bhr));
      float z  = sigmf((g6[1][r] + biz) + (g6[4][r] + bhz));
      float ghn = g6[5][r] + bhn;
      float nn = tanhf((g6[2][r] + bin) + rg * ghn);
      float hold = h[(size_t)row * HID + c];
      float hv = (1.f - z) * nn + z * hold;
      h[(size_t)row * HID + c] = hv;
      x16[(size_t)row * HID + c] = (_Float16)hv;
    }
  }
}

// ---------------- Set2Set: LSTM cell (gate order i,f,g,o) ----------------
__global__ __launch_bounds__(64) void lstm_kernel(
    const float* __restrict__ qstar, float* __restrict__ qh, float* __restrict__ qc,
    const float* __restrict__ Wih, const float* __restrict__ Whh,
    const float* __restrict__ bih, const float* __restrict__ bhh) {
  __shared__ float sq[2 * HID];
  __shared__ float sh_[HID];
  int b = blockIdx.x, c = threadIdx.x;
  sq[c] = qstar[b * 2 * HID + c];
  sq[HID + c] = qstar[b * 2 * HID + HID + c];
  sh_[c] = qh[b * HID + c];
  __syncthreads();
  float g[4];
#pragma unroll
  for (int gate = 0; gate < 4; ++gate) {
    int o = gate * HID + c;
    float s = bih[o] + bhh[o];
    const float* wi = Wih + o * 2 * HID;
    const float* wh = Whh + o * HID;
    for (int k = 0; k < 2 * HID; ++k) s += sq[k] * wi[k];
    for (int k = 0; k < HID; ++k) s += sh_[k] * wh[k];
    g[gate] = s;
  }
  float cprev = qc[b * HID + c];
  float cn = sigmf(g[1]) * cprev + sigmf(g[0]) * tanhf(g[2]);
  float hn = sigmf(g[3]) * tanhf(cn);
  qc[b * HID + c] = cn;
  qh[b * HID + c] = hn;
}

__global__ void attn_init_kernel(unsigned* emax_u, float* asum, float* rvec) {
  int i = blockIdx.x * 256 + threadIdx.x;
  if (i < NUM_GRAPHS * HID) rvec[i] = 0.f;
  if (i < NUM_GRAPHS) { emax_u[i] = 0x007FFFFFu; /* mapped -inf */ asum[i] = 0.f; }
}

__global__ void attn1_kernel(const float* __restrict__ h, const float* __restrict__ qh,
                             const int* __restrict__ batch, float* __restrict__ e_buf,
                             unsigned* __restrict__ emax_u, int nNodes) {
  int i = blockIdx.x * 256 + threadIdx.x;
  if (i >= nNodes) return;
  int g = batch[i];
  const float* xr = h + (size_t)i * HID;
  const float* q  = qh + (size_t)g * HID;
  float e = 0.f;
#pragma unroll 8
  for (int k = 0; k < HID; ++k) e += xr[k] * q[k];
  e_buf[i] = e;
  atomicMax(emax_u + g, fmapu(e));
}

__global__ void attn2_kernel(const float* __restrict__ h, const int* __restrict__ batch,
                             const float* __restrict__ e_buf, const unsigned* __restrict__ emax_u,
                             float* __restrict__ asum, float* __restrict__ rvec, int nNodes) {
  int idx = blockIdx.x * 256 + threadIdx.x;
  int i = idx >> 6, c = idx & 63;
  if (i >= nNodes) return;
  int g = batch[i];
  float a = expf(e_buf[i] - fumap(emax_u[g]));
  atomicAdd(rvec + (size_t)g * HID + c, a * h[(size_t)i * HID + c]);
  if (c == 0) atomicAdd(asum + g, a);
}

__global__ void qstar_kernel(const float* __restrict__ qh, const float* __restrict__ rvec,
                             const float* __restrict__ asum, float* __restrict__ qstar) {
  int idx = blockIdx.x * 256 + threadIdx.x;
  if (idx >= NUM_GRAPHS * HID) return;
  int b = idx >> 6, c = idx & 63;
  qstar[b * 2 * HID + c] = qh[idx];
  qstar[b * 2 * HID + HID + c] = rvec[idx] / asum[b];
}

// ---------------- regressor: Linear(2H->H) -> ReLU -> Linear(H->1) ----------------
__global__ __launch_bounds__(64) void reg_kernel(
    const float* __restrict__ qstar, const float* __restrict__ W1, const float* __restrict__ b1,
    const float* __restrict__ W2, const float* __restrict__ b2, float* __restrict__ out) {
  __shared__ float sq[2 * HID];
  __shared__ float sp[HID];
  int b = blockIdx.x, c = threadIdx.x;
  sq[c] = qstar[b * 2 * HID + c];
  sq[HID + c] = qstar[b * 2 * HID + HID + c];
  __syncthreads();
  float s = b1[c];
  for (int k = 0; k < 2 * HID; ++k) s += sq[k] * W1[k * HID + c];
  s = fmaxf(s, 0.f);
  sp[c] = s * W2[c];
  __syncthreads();
  for (int off = 32; off > 0; off >>= 1) {
    if (c < off) sp[c] += sp[c + off];
    __syncthreads();
  }
  if (c == 0) out[b] = sp[0] + b2[0];
}

// ---------------- driver ----------------
extern "C" void kernel_launch(void* const* d_in, const int* in_sizes, int n_in,
                              void* d_out, int out_size, void* d_ws, size_t ws_size,
                              hipStream_t stream) {
  (void)in_sizes; (void)n_in; (void)out_size; (void)ws_size;
  const float* x_feat = (const float*)d_in[0];
  const int*   eidx   = (const int*)d_in[1];
  const float* eattr  = (const float*)d_in[2];
  const int*   batch  = (const int*)d_in[3];
  const float* W_emb  = (const float*)d_in[4];
  const float* b_emb  = (const float*)d_in[5];
  const float* W_m1   = (const float*)d_in[6];
  const float* b_m1   = (const float*)d_in[7];
  const float* W_m2   = (const float*)d_in[8];
  const float* b_m2   = (const float*)d_in[9];
  const float* gWih   = (const float*)d_in[10];
  const float* gWhh   = (const float*)d_in[11];
  const float* gbih   = (const float*)d_in[12];
  const float* gbhh   = (const float*)d_in[13];
  const float* lWih   = (const float*)d_in[14];
  const float* lWhh   = (const float*)d_in[15];
  const float* lbih   = (const float*)d_in[16];
  const float* lbhh   = (const float*)d_in[17];
  const float* W_r1   = (const float*)d_in[18];
  const float* b_r1   = (const float*)d_in[19];
  const float* W_r2   = (const float*)d_in[20];
  const float* b_r2   = (const float*)d_in[21];
  float* out = (float*)d_out;

  const int* src = eidx;
  const int* dst = eidx + N_EDGES;

  // workspace carve-up (256-B aligned)
  char* wp = (char*)d_ws;
  auto alloc = [&](size_t bytes) -> void* {
    void* p = (void*)wp;
    wp += (bytes + 255) & ~(size_t)255;
    return p;
  };
  float*    h_buf  = (float*)alloc((size_t)N_NODES * HID * 4);
  float*    m_buf  = (float*)alloc((size_t)N_NODES * HID * 4);
  _Float16* x16    = (_Float16*)alloc((size_t)N_NODES * HID * 2);
  _Float16* m16    = (_Float16*)alloc((size_t)N_NODES * HID * 2);
  _Float16* ea16   = (_Float16*)alloc((size_t)N_EDGES * EDGE_DIM * 2);
  _Float16* W1p    = (_Float16*)alloc(5 * 8 * 32 * 16 * 2);
  _Float16* W2p    = (_Float16*)alloc(4 * 4 * 32 * 16 * 2);
  _Float16* Wihp   = (_Float16*)alloc(2 * 12 * 32 * 16 * 2);
  _Float16* Whhp   = (_Float16*)alloc(2 * 12 * 32 * 16 * 2);
  float*    qh     = (float*)alloc(NUM_GRAPHS * HID * 4);
  float*    qc     = (float*)alloc(NUM_GRAPHS * HID * 4);
  float*    qstar  = (float*)alloc(NUM_GRAPHS * 2 * HID * 4);
  float*    e_buf  = (float*)alloc((size_t)N_NODES * 4);
  unsigned* emax_u = (unsigned*)alloc(NUM_GRAPHS * 4);
  float*    asum   = (float*)alloc(NUM_GRAPHS * 4);
  float*    rvec   = (float*)alloc(NUM_GRAPHS * HID * 4);

  // one-time preprocessing (runs inside the captured graph; deterministic)
  pack_w1_kernel<<<(5 * 8 * 32 * 16 + 255) / 256, 256, 0, stream>>>(W_m1, W1p);
  pack_w2_kernel<<<(4 * 4 * 32 * 16 + 255) / 256, 256, 0, stream>>>(W_m2, W2p);
  pack_gru_kernel<<<(2 * 12 * 32 * 16 + 255) / 256, 256, 0, stream>>>(gWih, Wihp);
  pack_gru_kernel<<<(2 * 12 * 32 * 16 + 255) / 256, 256, 0, stream>>>(gWhh, Whhp);
  f32_to_f16_kernel<<<(N_EDGES * EDGE_DIM + 255) / 256, 256, 0, stream>>>(eattr, ea16, N_EDGES * EDGE_DIM);
  embed_kernel<<<N_NODES / 4, 256, 0, stream>>>(x_feat, W_emb, b_emb, h_buf, x16, N_NODES);

  // message passing rounds
  const int gruTiles = N_NODES / 16;                  // 3125 exactly
  for (int mp = 0; mp < NUM_MP; ++mp) {
    zero_kernel<<<(N_NODES * HID + 255) / 256, 256, 0, stream>>>(m_buf, N_NODES * HID);
    edge_mlp_kernel<<<N_EDGES / 64, 128, 0, stream>>>(x16, ea16, src, dst, W1p, W2p, b_m1, b_m2, m_buf);
    f32_to_f16_kernel<<<(N_NODES * HID + 255) / 256, 256, 0, stream>>>(m_buf, m16, N_NODES * HID);
    gru_wmma_kernel<<<(gruTiles + 3) / 4, 128, 0, stream>>>(m16, h_buf, x16, Wihp, Whhp, gbih, gbhh, gruTiles);
  }

  // Set2Set pooling
  zero_kernel<<<(NUM_GRAPHS * HID + 255) / 256, 256, 0, stream>>>(qh, NUM_GRAPHS * HID);
  zero_kernel<<<(NUM_GRAPHS * HID + 255) / 256, 256, 0, stream>>>(qc, NUM_GRAPHS * HID);
  zero_kernel<<<(NUM_GRAPHS * 2 * HID + 255) / 256, 256, 0, stream>>>(qstar, NUM_GRAPHS * 2 * HID);
  for (int s = 0; s < S2S_STEPS; ++s) {
    lstm_kernel<<<NUM_GRAPHS, 64, 0, stream>>>(qstar, qh, qc, lWih, lWhh, lbih, lbhh);
    attn_init_kernel<<<(NUM_GRAPHS * HID + 255) / 256, 256, 0, stream>>>(emax_u, asum, rvec);
    attn1_kernel<<<(N_NODES + 255) / 256, 256, 0, stream>>>(h_buf, qh, batch, e_buf, emax_u, N_NODES);
    attn2_kernel<<<(N_NODES * HID + 255) / 256, 256, 0, stream>>>(h_buf, batch, e_buf, emax_u, asum, rvec, N_NODES);
    qstar_kernel<<<(NUM_GRAPHS * HID + 255) / 256, 256, 0, stream>>>(qh, rvec, asum, qstar);
  }

  reg_kernel<<<NUM_GRAPHS, 64, 0, stream>>>(qstar, W_r1, b_r1, W_r2, b_r2, out);
}
